// EGCNet_55594056679488
// MI455X (gfx1250) — matrix operand order repaired and checked by reading, not compiled, not run on hardware
//
#include <hip/hip_runtime.h>
#include <math.h>

typedef __attribute__((ext_vector_type(16))) _Float16 v16h;
typedef __attribute__((ext_vector_type(8)))  _Float16 v8h;
typedef __attribute__((ext_vector_type(8)))  float    v8f;
typedef __attribute__((ext_vector_type(4)))  unsigned int v4u;
typedef __attribute__((ext_vector_type(8)))  int v8i;
typedef __attribute__((ext_vector_type(4)))  int v4i;

static constexpr int kN   = 50000;
static constexpr int kE   = 800000;
static constexpr int kHid = 128;
static constexpr int kNodeTiles = kN / 16;   // 3125
static constexpr int kEdgeTiles = kE / 16;   // 50000
static constexpr int kWaves = 8;             // waves per 256-thread block

#if defined(__has_builtin)
#  if __has_builtin(__builtin_amdgcn_tensor_load_to_lds) && __has_builtin(__builtin_amdgcn_s_wait_tensorcnt)
#    define USE_TDM 1
#  endif
#endif
#ifndef USE_TDM
#  define USE_TDM 0
#endif

__device__ __forceinline__ float sigmoid_(float x){ return 1.0f/(1.0f+__expf(-x)); }
__device__ __forceinline__ float silu_(float x){ return x*sigmoid_(x); }

// A operand (16x32 f16, M x K). lane<16 -> row m=lane, K {base..base+7} u {base+16..base+23},
// base = (lane>=16)*8.  p points at (row start + kchunk*32) in a row-major f16 buffer.
__device__ __forceinline__ v16h load_a16(const _Float16* p, int hi){
  const _Float16* q = p + hi*8;
  v8h lo = *(const v8h*)q;
  v8h hh = *(const v8h*)(q+16);
  v16h a;
#pragma unroll
  for (int j=0;j<8;++j){ a[j]=lo[j]; a[j+8]=hh[j]; }
  return a;
}

// B operand (32x16 f16, K x N). col n=lane%16, K {base..base+7} u {base+16..base+23}, base=(lane>=16)*8.
// Weights stored transposed WT[n][k] row-major -> two runs of 8 contiguous halfs (k and k+16).
__device__ __forceinline__ v16h load_b16(const _Float16* p, int hi){
  const _Float16* q = p + hi*8;
  v8h b0 = *(const v8h*)q;
  v8h b1 = *(const v8h*)(q+16);
  v16h b;
#pragma unroll
  for (int j=0;j<8;++j){ b[j]=b0[j]; b[j+8]=b1[j]; }
  return b;
}

__device__ __forceinline__ v8f wmma16(v16h a, v16h b, v8f c){
  return __builtin_amdgcn_wmma_f32_16x16x32_f16(false, a, false, b, (short)0, c, false, false);
}
__device__ __forceinline__ v8f vzero8(){ v8f z = {0.f,0.f,0.f,0.f,0.f,0.f,0.f,0.f}; return z; }

#if USE_TDM
// Issue one TDM load of `nelem` f16 elements (1-D tile) from global to LDS.
// D# layout per CDNA5 ISA ch.8: group0 {flags, lds_addr, global_addr, type=2},
// group1 {data_size=2B, tensor_dim0=tile_dim0=nelem, dim1=1, strides}.
// This toolchain exposes the 6-arg builtin: (g0 v4u, g1 v8i, g2 v4i, g3 v4i, gx v8i, cpol i32).
__device__ __forceinline__ void tdm_load_f16_1d(unsigned lds_byte_addr,
                                                const _Float16* gsrc,
                                                unsigned nelem){
  unsigned long long ga = (unsigned long long)(const void*)gsrc;
  v4u g0;
  g0[0] = 1u;                                                  // count=1, user descriptor
  g0[1] = lds_byte_addr;                                       // lds_addr
  g0[2] = (unsigned)ga;                                        // global_addr[31:0]
  g0[3] = (unsigned)((ga >> 32) & 0x01FFFFFFu) | (2u << 30);   // global_addr[56:32] | type=2
  v8i g1;
  g1[0] = (int)(1u << 16);                                     // data_size=1 -> 2 bytes
  g1[1] = (int)((nelem & 0xFFFFu) << 16);                      // tensor_dim0[15:0]
  g1[2] = (int)(((nelem >> 16) & 0xFFFFu) | (1u << 16));       // tensor_dim0[31:16] | tensor_dim1=1
  g1[3] = (int)((nelem & 0xFFFFu) << 16);                      // tensor_dim1[31:16]=0 | tile_dim0
  g1[4] = 1;                                                   // tile_dim1=1, tile_dim2=0
  g1[5] = (int)nelem;                                          // tensor_dim0_stride[31:0]
  g1[6] = (int)((nelem & 0xFFFFu) << 16);                      // stride0[47:32]=0 | dim1_stride[15:0]
  g1[7] = (int)((nelem >> 16) & 0xFFFFu);                      // dim1_stride[47:16]
  v4i g2 = {0,0,0,0};
  v4i g3 = {0,0,0,0};
  v8i gx = {0,0,0,0,0,0,0,0};
  __builtin_amdgcn_tensor_load_to_lds(g0, g1, g2, g3, gx, 0);
}
__device__ __forceinline__ unsigned lds_addr_of(const void* p){
  return (unsigned)(unsigned long long)p;    // low 32 bits of generic address = LDS offset
}
#endif

// ---------------------------------------------------------------------------
// Weight prep: transpose + convert to f16.  WT[n][k] = W[k][n].
// ---------------------------------------------------------------------------
__global__ void prep_weights_kernel(const float* __restrict__ Wemb, const float* __restrict__ We1,
                                    const float* __restrict__ We2,  const float* __restrict__ Wh1,
                                    const float* __restrict__ Wh2,  const float* __restrict__ Wl1,
                                    _Float16* __restrict__ WembT, _Float16* __restrict__ We1T,
                                    _Float16* __restrict__ We2T,  _Float16* __restrict__ Wh1T,
                                    _Float16* __restrict__ Wh2T,  _Float16* __restrict__ Wl1T)
{
  int i = blockIdx.x*blockDim.x + threadIdx.x;
  if (i < 128*32)  { int n=i/32,  k=i%32;  WembT[i]=(_Float16)Wemb[k*128+n]; return; } i -= 128*32;
  if (i < 128*256) { int n=i/256, k=i%256; We1T[i]=(_Float16)We1[k*128+n];   return; } i -= 128*256;
  if (i < 128*128) { int n=i/128, k=i%128; We2T[i]=(_Float16)We2[k*128+n];   return; } i -= 128*128;
  if (i < 128*256) { int n=i/256, k=i%256; Wh1T[i]=(_Float16)Wh1[k*128+n];   return; } i -= 128*256;
  if (i < 128*128) { int n=i/128, k=i%128; Wh2T[i]=(_Float16)Wh2[k*128+n];   return; } i -= 128*128;
  if (i < 128*128) { int n=i/128, k=i%128; Wl1T[i]=(_Float16)Wl1[k*128+n];   return; }
}

__global__ void cvt_f32_to_f16_kernel(const float* __restrict__ src, _Float16* __restrict__ dst, int n)
{
  int i = blockIdx.x*blockDim.x + threadIdx.x;
  if (i < n) dst[i] = (_Float16)src[i];
}

// ---------------------------------------------------------------------------
// Embedding: h = x @ W_emb + b_emb.
// ---------------------------------------------------------------------------
__launch_bounds__(256)
__global__ void embed_kernel(const float* __restrict__ x, const _Float16* __restrict__ WembT,
                             const float* __restrict__ b_emb,
                             float* __restrict__ hf, _Float16* __restrict__ h16)
{
  const int lane = threadIdx.x & 31;
  const int w    = threadIdx.x >> 5;
  const int lo   = lane & 15, hi = lane >> 4;
  const int tile = blockIdx.x*kWaves + w;
  if (tile >= kNodeTiles) return;
  const int node_m = tile*16 + lo;

  const float* px = x + (size_t)node_m*32;
  const float* p0 = px + hi*8;
  const float* p1 = px + 16 + hi*8;
  v16h a;
#pragma unroll
  for (int j=0;j<8;++j){ a[j]=(_Float16)p0[j]; a[j+8]=(_Float16)p1[j]; }

#pragma unroll
  for (int n=0;n<8;++n){
    v16h b = load_b16(WembT + (size_t)(n*16+lo)*32, hi);
    v8f c = vzero8();
    c = wmma16(a,b,c);
    const int col = n*16 + lo;
    const float be = b_emb[col];
#pragma unroll
    for (int r=0;r<8;++r){
      const size_t node = (size_t)(tile*16 + r + 8*hi);
      float hv = c[r] + be;
      hf [node*kHid + col] = hv;
      h16[node*kHid + col] = (_Float16)hv;
    }
  }
}

// ---------------------------------------------------------------------------
// Fused edge MLP + gate + segment-sum scatter.  One wave per 16 edges.
// Weights staged once per block into LDS (TDM if available).
// ---------------------------------------------------------------------------
__launch_bounds__(256)
__global__ void edge_mlp_kernel(const float* __restrict__ x, const long long* __restrict__ eidx,
                                const _Float16* __restrict__ h16,
                                const _Float16* __restrict__ W1T, const _Float16* __restrict__ W2T,
                                const float* __restrict__ We1,   // full [257][128] (for dist row 256)
                                const float* __restrict__ b_e1, const float* __restrict__ b_e2,
                                const float* __restrict__ Winf, const float* __restrict__ binf,
                                float* __restrict__ mi)
{
  __shared__ alignas(16) _Float16 sh_w1[128*256];     // 64 KB
  __shared__ alignas(16) _Float16 sh_w2[128*128];     // 32 KB
  __shared__ alignas(16) _Float16 sh_tile[kWaves][16*kHid];  // 32 KB
  __shared__ unsigned sh_s[kWaves][16];
  __shared__ unsigned sh_t[kWaves][16];
  __shared__ float    sh_d[kWaves][16];

  const int lane = threadIdx.x & 31;
  const int w    = threadIdx.x >> 5;
  const int lo   = lane & 15, hi = lane >> 4;
  const int tile = blockIdx.x*kWaves + w;        // grid sized exactly: kEdgeTiles
  const int e0   = tile*16;

  // ---- stage weights into LDS
#if USE_TDM
  if (threadIdx.x < 32){
    tdm_load_f16_1d(lds_addr_of(sh_w1), W1T, 128*256);
    tdm_load_f16_1d(lds_addr_of(sh_w2), W2T, 128*128);
    __builtin_amdgcn_s_wait_tensorcnt(0);
  }
#else
  for (int i = threadIdx.x; i < (128*256)/8; i += 256)
    ((v8h*)sh_w1)[i] = ((const v8h*)W1T)[i];
  for (int i = threadIdx.x; i < (128*128)/8; i += 256)
    ((v8h*)sh_w2)[i] = ((const v8h*)W2T)[i];
#endif

  if (lane < 16){
    long long s = eidx[e0 + lane];
    long long t = eidx[(size_t)kE + e0 + lane];
    float dx = x[(size_t)s*32 + 0] - x[(size_t)t*32 + 0];
    float dy = x[(size_t)s*32 + 1] - x[(size_t)t*32 + 1];
    sh_s[w][lane] = (unsigned)s;
    sh_t[w][lane] = (unsigned)t;
    sh_d[w][lane] = sqrtf(dx*dx + dy*dy);
  }
  __syncthreads();

  const unsigned rs = sh_s[w][lo];
  const unsigned rt = sh_t[w][lo];

  // ---- layer 1: [16 x 257] @ [257 x 128]; K=0..255 via WMMA, dist row via rank-1 update
  v8f acc[8];
#pragma unroll
  for (int n=0;n<8;++n) acc[n] = vzero8();

#pragma unroll
  for (int kc=0; kc<8; ++kc){
    const unsigned row = (kc < 4) ? rs : rt;
    const int koff = (kc & 3)*32;
    v16h a = load_a16(h16 + (size_t)row*kHid + koff, hi);
#pragma unroll
    for (int n=0;n<8;++n){
      v16h b = load_b16(sh_w1 + (n*16+lo)*256 + kc*32, hi);
      acc[n] = wmma16(a,b,acc[n]);
    }
  }

  float dv[8];
#pragma unroll
  for (int r=0;r<8;++r) dv[r] = sh_d[w][r + 8*hi];

#pragma unroll
  for (int n=0;n<8;++n){
    const int col = n*16 + lo;
    const float be = b_e1[col];
    const float wr = We1[256*kHid + col];      // dist weight row
#pragma unroll
    for (int r=0;r<8;++r){
      float v = acc[n][r] + be + dv[r]*wr;
      v = silu_(v);
      sh_tile[w][(r + 8*hi)*kHid + col] = (_Float16)v;
    }
  }
  __syncthreads();

  // ---- layer 2: [16 x 128] @ [128 x 128]
  v8f acc2[8];
#pragma unroll
  for (int n=0;n<8;++n) acc2[n] = vzero8();

#pragma unroll
  for (int kc=0; kc<4; ++kc){
    v16h a = load_a16(&sh_tile[w][(size_t)lo*kHid + kc*32], hi);
#pragma unroll
    for (int n=0;n<8;++n){
      v16h b = load_b16(sh_w2 + (n*16+lo)*kHid + kc*32, hi);
      acc2[n] = wmma16(a,b,acc2[n]);
    }
  }

  // m_ij = silu(. + b_e2); gate e_ij = sigmoid(m_ij . w_inf + b_inf)
  float part[8];
#pragma unroll
  for (int r=0;r<8;++r) part[r] = 0.f;

#pragma unroll
  for (int n=0;n<8;++n){
    const int col = n*16 + lo;
    const float be = b_e2[col];
    const float wi = Winf[col];
#pragma unroll
    for (int r=0;r<8;++r){
      float v = silu_(acc2[n][r] + be);
      acc2[n][r] = v;                // acc2 now holds m_ij
      part[r] += v * wi;
    }
  }

  const float bi = binf[0];
  float e[8];
#pragma unroll
  for (int r=0;r<8;++r){
    float v = part[r];
    v += __shfl_xor(v, 8, 16);
    v += __shfl_xor(v, 4, 16);
    v += __shfl_xor(v, 2, 16);
    v += __shfl_xor(v, 1, 16);
    e[r] = sigmoid_(v + bi);
  }

  unsigned srow[8];
#pragma unroll
  for (int r=0;r<8;++r) srow[r] = sh_s[w][r + 8*hi];

#pragma unroll
  for (int n=0;n<8;++n){
    const int col = n*16 + lo;
#pragma unroll
    for (int r=0;r<8;++r){
      unsafeAtomicAdd(&mi[(size_t)srow[r]*kHid + col], e[r]*acc2[n][r]);
    }
  }
}

// ---------------------------------------------------------------------------
// Fused node update + readout.  One wave per 16 nodes.
// ---------------------------------------------------------------------------
__launch_bounds__(256)
__global__ void node_update_kernel(const _Float16* __restrict__ h16, const _Float16* __restrict__ mi16,
                                   const float* __restrict__ hf,
                                   const _Float16* __restrict__ Wh1T, const _Float16* __restrict__ Wh2T,
                                   const _Float16* __restrict__ Wl1T,
                                   const float* __restrict__ b_h1, const float* __restrict__ b_h2,
                                   const float* __restrict__ b_l1,
                                   const float* __restrict__ Wl2, const float* __restrict__ b_l2,
                                   float* __restrict__ out)
{
  __shared__ alignas(16) _Float16 sh_tile[kWaves][16*kHid];

  const int lane = threadIdx.x & 31;
  const int w    = threadIdx.x >> 5;
  const int lo   = lane & 15, hi = lane >> 4;
  const int tile0 = blockIdx.x*kWaves + w;
  const bool active = (tile0 < kNodeTiles);
  const int tile = active ? tile0 : 0;            // inactive waves still run (barriers!)
  const int node0 = tile*16;

  // ---- hm @ W_h1 : K=256 (h | m_i)
  v8f acc[8];
#pragma unroll
  for (int n=0;n<8;++n) acc[n] = vzero8();

#pragma unroll
  for (int kc=0; kc<8; ++kc){
    const _Float16* src = (kc < 4) ? h16 : mi16;
    const int koff = (kc & 3)*32;
    v16h a = load_a16(src + (size_t)(node0 + lo)*kHid + koff, hi);
#pragma unroll
    for (int n=0;n<8;++n){
      v16h b = load_b16(Wh1T + (size_t)(n*16+lo)*256 + kc*32, hi);
      acc[n] = wmma16(a,b,acc[n]);
    }
  }

#pragma unroll
  for (int n=0;n<8;++n){
    const int col = n*16 + lo;
    const float be = b_h1[col];
#pragma unroll
    for (int r=0;r<8;++r)
      sh_tile[w][(r + 8*hi)*kHid + col] = (_Float16)silu_(acc[n][r] + be);
  }
  __syncthreads();

  // ---- @ W_h2, + b_h2 + residual h
  v8f acc2[8];
#pragma unroll
  for (int n=0;n<8;++n) acc2[n] = vzero8();
#pragma unroll
  for (int kc=0; kc<4; ++kc){
    v16h a = load_a16(&sh_tile[w][(size_t)lo*kHid + kc*32], hi);
#pragma unroll
    for (int n=0;n<8;++n){
      v16h b = load_b16(Wh2T + (size_t)(n*16+lo)*kHid + kc*32, hi);
      acc2[n] = wmma16(a,b,acc2[n]);
    }
  }
#pragma unroll
  for (int n=0;n<8;++n){
    const int col = n*16 + lo;
    const float be = b_h2[col];
#pragma unroll
    for (int r=0;r<8;++r){
      const size_t node = (size_t)(node0 + r + 8*hi);
      acc2[n][r] = hf[node*kHid + col] + acc2[n][r] + be;   // h2
    }
  }
  __syncthreads();                                  // tile reads done, safe to overwrite
#pragma unroll
  for (int n=0;n<8;++n){
    const int col = n*16 + lo;
#pragma unroll
    for (int r=0;r<8;++r)
      sh_tile[w][(r + 8*hi)*kHid + col] = (_Float16)acc2[n][r];
  }
  __syncthreads();

  // ---- tanh(h2 @ W_l1 + b_l1)
  v8f acc3[8];
#pragma unroll
  for (int n=0;n<8;++n) acc3[n] = vzero8();
#pragma unroll
  for (int kc=0; kc<4; ++kc){
    v16h a = load_a16(&sh_tile[w][(size_t)lo*kHid + kc*32], hi);
#pragma unroll
    for (int n=0;n<8;++n){
      v16h b = load_b16(Wl1T + (size_t)(n*16+lo)*kHid + kc*32, hi);
      acc3[n] = wmma16(a,b,acc3[n]);
    }
  }
  __syncthreads();
#pragma unroll
  for (int n=0;n<8;++n){
    const int col = n*16 + lo;
    const float be = b_l1[col];
#pragma unroll
    for (int r=0;r<8;++r)
      sh_tile[w][(r + 8*hi)*kHid + col] = (_Float16)tanhf(acc3[n][r] + be);
  }
  __syncthreads();

  // ---- @ W_l2 (128 x 3) : per-row dot products
  if (lane < 16 && active){
    const int node = node0 + lane;
    float s0 = b_l2[0], s1 = b_l2[1], s2 = b_l2[2];
#pragma unroll 8
    for (int k=0;k<kHid;++k){
      float v = (float)sh_tile[w][lane*kHid + k];
      s0 += v*Wl2[k*3+0];
      s1 += v*Wl2[k*3+1];
      s2 += v*Wl2[k*3+2];
    }
    out[(size_t)node*3 + 0] = s0;
    out[(size_t)node*3 + 1] = s1;
    out[(size_t)node*3 + 2] = s2;
  }
}

// ---------------------------------------------------------------------------
extern "C" void kernel_launch(void* const* d_in, const int* in_sizes, int n_in,
                              void* d_out, int out_size, void* d_ws, size_t ws_size,
                              hipStream_t stream)
{
  (void)in_sizes; (void)n_in; (void)out_size; (void)ws_size;

  const float*     x    = (const float*)d_in[0];
  const long long* eidx = (const long long*)d_in[1];
  const float* Wemb = (const float*)d_in[2];  const float* b_emb = (const float*)d_in[3];
  const float* We1  = (const float*)d_in[4];  const float* b_e1  = (const float*)d_in[5];
  const float* We2  = (const float*)d_in[6];  const float* b_e2  = (const float*)d_in[7];
  const float* Winf = (const float*)d_in[8];  const float* binf  = (const float*)d_in[9];
  const float* Wh1  = (const float*)d_in[10]; const float* b_h1  = (const float*)d_in[11];
  const float* Wh2  = (const float*)d_in[12]; const float* b_h2  = (const float*)d_in[13];
  const float* Wl1  = (const float*)d_in[14]; const float* b_l1  = (const float*)d_in[15];
  const float* Wl2  = (const float*)d_in[16]; const float* b_l2  = (const float*)d_in[17];
  float* out = (float*)d_out;

  char* ws = (char*)d_ws;
  size_t off = 0;
  auto alloc = [&](size_t bytes)->void* {
    void* p = (void*)(ws + off);
    off += (bytes + 255) & ~(size_t)255;
    return p;
  };
  float*     hf    = (float*)    alloc((size_t)kN*kHid*sizeof(float));
  _Float16*  h16   = (_Float16*) alloc((size_t)kN*kHid*sizeof(_Float16));
  float*     mi    = (float*)    alloc((size_t)kN*kHid*sizeof(float));
  _Float16*  mi16  = (_Float16*) alloc((size_t)kN*kHid*sizeof(_Float16));
  _Float16*  WembT = (_Float16*) alloc(128*32*sizeof(_Float16));
  _Float16*  We1T  = (_Float16*) alloc(128*256*sizeof(_Float16));
  _Float16*  We2T  = (_Float16*) alloc(128*128*sizeof(_Float16));
  _Float16*  Wh1T  = (_Float16*) alloc(128*256*sizeof(_Float16));
  _Float16*  Wh2T  = (_Float16*) alloc(128*128*sizeof(_Float16));
  _Float16*  Wl1T  = (_Float16*) alloc(128*128*sizeof(_Float16));

  (void)hipMemsetAsync(mi, 0, (size_t)kN*kHid*sizeof(float), stream);

  const int prepElems = 128*(32 + 256 + 128 + 256 + 128 + 128);
  prep_weights_kernel<<<(prepElems + 255)/256, 256, 0, stream>>>(
      Wemb, We1, We2, Wh1, Wh2, Wl1, WembT, We1T, We2T, Wh1T, Wh2T, Wl1T);

  const int nodeBlocks = (kNodeTiles + kWaves - 1)/kWaves;   // 391
  embed_kernel<<<nodeBlocks, 256, 0, stream>>>(x, WembT, b_emb, hf, h16);

  edge_mlp_kernel<<<kEdgeTiles/kWaves, 256, 0, stream>>>(
      x, eidx, h16, We1T, We2T, We1, b_e1, b_e2, Winf, binf, mi);

  cvt_f32_to_f16_kernel<<<((kN*kHid) + 255)/256, 256, 0, stream>>>(mi, mi16, kN*kHid);

  node_update_kernel<<<nodeBlocks, 256, 0, stream>>>(
      h16, mi16, hf, Wh1T, Wh2T, Wl1T, b_h1, b_h2, b_l1, Wl2, b_l2, out);
}
